// GraphConvolution_45045617001162
// MI455X (gfx1250) — compile-verified, matrix operand order checked by reading, not run on hardware
//
#include <hip/hip_runtime.h>

typedef __attribute__((ext_vector_type(16))) _Float16 v16h;
typedef __attribute__((ext_vector_type(8)))  _Float16 v8h;
typedef __attribute__((ext_vector_type(4)))  _Float16 v4h;
typedef __attribute__((ext_vector_type(8)))  float    v8f;

#define D_    64
#define DE_   16
#define H_    128
#define DOUT_ 64
#define KP_   160   // 2*D + DE = 144 padded to 5 K-tiles of 32
#define WPB   8     // waves per block (256 threads)

// A/B fragment K index for 16-bit 16x32 WMMA operands (ISA 7.12.2):
// lane half 'hi' (0: lanes0-15, 1: lanes16-31), VGPR v, h in {0,1}:
//   K = (v>=4 ? 16:0) + (hi ? 8:0) + 2*(v&3) + h
__device__ __forceinline__ int kidx0(int v, int hi) {
  return ((v & 4) << 2) + (hi << 3) + ((v & 3) << 1);
}

__device__ __forceinline__ float silu_f(float x) {
  return x / (1.f + __expf(-x));
}

__device__ __forceinline__ v16h cat8(v8h lo, v8h hi) {
  return __builtin_shufflevector(lo, hi, 0, 1, 2, 3, 4, 5, 6, 7,
                                 8, 9, 10, 11, 12, 13, 14, 15);
}

// Multiply one 16x32 A fragment against NTILES 16-col B tiles taken from an
// LDS weight matrix stored transposed as [n][k] with row stride STRIDE.
template <int NTILES, int STRIDE>
__device__ __forceinline__ void mma_bcast(const v16h a, int kbase, int m, int hi,
                                          const _Float16* __restrict__ sw,
                                          v8f* __restrict__ acc) {
#pragma unroll
  for (int j = 0; j < NTILES; ++j) {
    v16h b;
    const int n = j * 16 + m;
#pragma unroll
    for (int v = 0; v < 8; ++v) {
      const int k = kbase + kidx0(v, hi);
      b[2 * v]     = sw[n * STRIDE + k];
      b[2 * v + 1] = sw[n * STRIDE + k + 1];
    }
    acc[j] = __builtin_amdgcn_wmma_f32_16x16x32_f16(
        false, a, false, b, (short)0, acc[j], false, false);
  }
}

// ---------------------------------------------------------------------------
// Prep: node_features f32 -> f16 copy (each node row is re-gathered ~16x by
// the edge kernel, so one conversion pass halves gather bytes and kills the
// per-edge v_cvt chains).
// ---------------------------------------------------------------------------
__global__ __launch_bounds__(256) void cvt_f16_kernel(
    const float* __restrict__ in, _Float16* __restrict__ outp, int total) {
  const int i = (blockIdx.x * blockDim.x + threadIdx.x) * 4;
  if (i + 3 < total) {
    const float4 p = *(const float4*)(in + i);
    v4h o;
    o[0] = (_Float16)p.x; o[1] = (_Float16)p.y;
    o[2] = (_Float16)p.z; o[3] = (_Float16)p.w;
    *(v4h*)(outp + i) = o;
  } else {
    for (int k = i; k < total; ++k) outp[k] = (_Float16)in[k];
  }
}

// ---------------------------------------------------------------------------
// Edge kernel: per 16-edge tile per wave:
//   A = [h_i | h_j | e_feat | 0pad] (16 x 160) f16, gathered from f16 copy
//   C1 = silu(A @ We1 + be1)  (16 x 128)   -> staged in LDS as f16
//   C2 = silu(C1 @ We2 + be2) (16 x 64)
//   attn = sigmoid(|h_i-h_j| @ Wa + ba)  fused into the h_i gathers; the h_j
//   halves loaded for attention are reused as the kt=2,3 A fragments.
//   atomicAdd(agg[first[e]], attn * C2)
// ---------------------------------------------------------------------------
__global__ __launch_bounds__(256) void edge_mlp_kernel(
    const _Float16* __restrict__ nfh, const long long* __restrict__ eidx,
    const float* __restrict__ ef,
    const float* __restrict__ We1, const float* __restrict__ be1,
    const float* __restrict__ We2, const float* __restrict__ be2,
    const float* __restrict__ Wa, const float* __restrict__ ba,
    float* __restrict__ agg, int Nn, int E) {
  __shared__ _Float16 sW1[H_][KP_];      // We1^T  [n][k], zero-padded K
  __shared__ _Float16 sW2[DOUT_][H_];    // We2^T  [n][k]
  __shared__ float    sB1[H_];
  __shared__ float    sB2[DOUT_];
  __shared__ float    sWa[D_];
  __shared__ _Float16 sStage[WPB][16][H_ + 4];

  const int tid = threadIdx.x;
  for (int i = tid; i < H_ * KP_; i += blockDim.x) {
    const int n = i / KP_, k = i % KP_;
    sW1[n][k] = (_Float16)((k < 2 * D_ + DE_) ? We1[k * H_ + n] : 0.f);
  }
  for (int i = tid; i < DOUT_ * H_; i += blockDim.x) {
    const int n = i / H_, k = i % H_;
    sW2[n][k] = (_Float16)We2[k * DOUT_ + n];
  }
  if (tid < H_)    sB1[tid] = be1[tid];
  if (tid < DOUT_) sB2[tid] = be2[tid];
  if (tid < D_)    sWa[tid] = Wa[tid];
  __syncthreads();

  const float baV  = ba[0];
  const int   lane = tid & 31;
  const int   m    = lane & 15;
  const int   hi   = lane >> 4;
  const int   wid  = tid >> 5;

  const int numTiles = (E + 15) >> 4;
  for (int t = blockIdx.x * WPB + wid; t < numTiles; t += gridDim.x * WPB) {
    const int edge = t * 16 + m;
    const int ec   = (edge < E) ? edge : 0;
    const int src  = (int)eidx[ec];
    const int dst  = (int)eidx[(size_t)E + ec];
    const _Float16* rowI = nfh + (size_t)src * D_;
    const _Float16* rowJ = nfh + (size_t)dst * D_;
    const float*    rowE = ef + (size_t)ec * DE_;

    v8f   acc[8] = {};
    float apart  = 0.f;
    v8h   jSave[4];

    // K-tiles 0,1: h_i columns; fuse attention |h_i - h_j| . Wa and keep the
    // h_j halves (identical per-lane feature subset) for K-tiles 2,3.
#pragma unroll
    for (int kt = 0; kt < 2; ++kt) {
      const int base = kt * 32 + hi * 8;
      const v8h iLo = *(const v8h*)(rowI + base);
      const v8h iHi = *(const v8h*)(rowI + base + 16);
      const v8h jLo = *(const v8h*)(rowJ + base);
      const v8h jHi = *(const v8h*)(rowJ + base + 16);
      jSave[2 * kt]     = jLo;
      jSave[2 * kt + 1] = jHi;
#pragma unroll
      for (int i = 0; i < 8; ++i) {
        apart += fabsf((float)iLo[i] - (float)jLo[i]) * sWa[base + i];
        apart += fabsf((float)iHi[i] - (float)jHi[i]) * sWa[base + 16 + i];
      }
      mma_bcast<8, KP_>(cat8(iLo, iHi), kt * 32, m, hi, &sW1[0][0], acc);
    }
    // K-tiles 2,3: h_j columns (reuse registers, no second gather)
#pragma unroll
    for (int kt = 2; kt < 4; ++kt) {
      mma_bcast<8, KP_>(cat8(jSave[2 * (kt - 2)], jSave[2 * (kt - 2) + 1]),
                        kt * 32, m, hi, &sW1[0][0], acc);
    }
    // K-tile 4: edge features (cols 128..143, f32 stream) + zero pad
    {
      const float4 e0 = *(const float4*)(rowE + hi * 8);
      const float4 e1 = *(const float4*)(rowE + hi * 8 + 4);
      v16h a;
      a[0] = (_Float16)e0.x; a[1] = (_Float16)e0.y;
      a[2] = (_Float16)e0.z; a[3] = (_Float16)e0.w;
      a[4] = (_Float16)e1.x; a[5] = (_Float16)e1.y;
      a[6] = (_Float16)e1.z; a[7] = (_Float16)e1.w;
#pragma unroll
      for (int i = 8; i < 16; ++i) a[i] = (_Float16)0.f;
      mma_bcast<8, KP_>(a, 128, m, hi, &sW1[0][0], acc);
    }

    const float afull = apart + __shfl_xor(apart, 16, 32);
    const float attn  = 1.f / (1.f + __expf(-(afull + baV)));

    // bias + SiLU, stage C1 as f16
#pragma unroll
    for (int j = 0; j < 8; ++j) {
      const int col = j * 16 + m;
      const float bb = sB1[col];
#pragma unroll
      for (int r = 0; r < 8; ++r)
        sStage[wid][r + 8 * hi][col] = (_Float16)silu_f(acc[j][r] + bb);
    }
    __builtin_amdgcn_wave_barrier();
    asm volatile("" ::: "memory");  // LDS is in-order per wave; block reordering

    // layer 2: C1 (16x128) @ We2 (128x64)
    v8f acc2[4] = {};
#pragma unroll
    for (int kt = 0; kt < 4; ++kt) {
      v16h a;
#pragma unroll
      for (int v = 0; v < 8; ++v) {
        const int k = kt * 32 + kidx0(v, hi);
        a[2 * v]     = sStage[wid][m][k];
        a[2 * v + 1] = sStage[wid][m][k + 1];
      }
      mma_bcast<4, H_>(a, kt * 32, m, hi, &sW2[0][0], acc2);
    }

    // bias + SiLU + attention gate + scatter-add to agg[first[e]]
    const int baseEdge = t * 16;
#pragma unroll
    for (int r = 0; r < 8; ++r) {
      const int   rowM = r + 8 * hi;               // edge row in tile
      const float aM   = __shfl(attn, rowM, 32);   // lanes 0..15 hold rows 0..15
      const int   sM   = __shfl(src, rowM, 32);
      const bool  ok   = (baseEdge + rowM) < E;
#pragma unroll
      for (int j = 0; j < 4; ++j) {
        const int col = j * 16 + m;
        const float x = silu_f(acc2[j][r] + sB2[col]) * aM;
        if (ok) atomicAdd(agg + (size_t)sM * DOUT_ + col, x);
      }
    }
  }
}

// ---------------------------------------------------------------------------
// Node kernel: per 16-node tile per wave:
//   A = [h (f16 copy) | agg (f32)] (16 x 128),
//   out = silu(A@Wn1+bn1) @ Wn2 + bn2
// ---------------------------------------------------------------------------
__global__ __launch_bounds__(256) void node_mlp_kernel(
    const _Float16* __restrict__ nfh, const float* __restrict__ agg,
    const float* __restrict__ Wn1, const float* __restrict__ bn1,
    const float* __restrict__ Wn2, const float* __restrict__ bn2,
    float* __restrict__ out, int Nn) {
  __shared__ _Float16 sW1[H_][H_];     // Wn1^T [n][k]
  __shared__ _Float16 sW2[DOUT_][H_];  // Wn2^T [n][k]
  __shared__ float    sB1[H_];
  __shared__ float    sB2[DOUT_];
  __shared__ _Float16 sStage[WPB][16][H_ + 4];

  const int tid = threadIdx.x;
  for (int i = tid; i < H_ * H_; i += blockDim.x) {
    const int n = i / H_, k = i % H_;
    sW1[n][k] = (_Float16)Wn1[k * H_ + n];
  }
  for (int i = tid; i < DOUT_ * H_; i += blockDim.x) {
    const int n = i / H_, k = i % H_;
    sW2[n][k] = (_Float16)Wn2[k * DOUT_ + n];
  }
  if (tid < H_)    sB1[tid] = bn1[tid];
  if (tid < DOUT_) sB2[tid] = bn2[tid];
  __syncthreads();

  const int lane = tid & 31, m = lane & 15, hi = lane >> 4, wid = tid >> 5;
  const int numTiles = (Nn + 15) >> 4;
  for (int t = blockIdx.x * WPB + wid; t < numTiles; t += gridDim.x * WPB) {
    const int node = t * 16 + m;
    const int nc   = (node < Nn) ? node : 0;
    const _Float16* rowH = nfh + (size_t)nc * D_;
    const float*    rowA = agg + (size_t)nc * DOUT_;

    v8f acc[8] = {};
    // K-tiles 0,1: node features (f16, direct fragment loads)
#pragma unroll
    for (int kt = 0; kt < 2; ++kt) {
      const int base = kt * 32 + hi * 8;
      const v8h lo = *(const v8h*)(rowH + base);
      const v8h hi8 = *(const v8h*)(rowH + base + 16);
      mma_bcast<8, H_>(cat8(lo, hi8), kt * 32, m, hi, &sW1[0][0], acc);
    }
    // K-tiles 2,3: aggregated messages (f32 -> f16)
#pragma unroll
    for (int kt = 2; kt < 4; ++kt) {
      v16h a;
#pragma unroll
      for (int v = 0; v < 8; ++v) {
        const int f = (kt - 2) * 32 + kidx0(v, hi);
        const float2 p = *(const float2*)(rowA + f);
        a[2 * v]     = (_Float16)p.x;
        a[2 * v + 1] = (_Float16)p.y;
      }
      mma_bcast<8, H_>(a, kt * 32, m, hi, &sW1[0][0], acc);
    }

#pragma unroll
    for (int j = 0; j < 8; ++j) {
      const int col = j * 16 + m;
      const float bb = sB1[col];
#pragma unroll
      for (int r = 0; r < 8; ++r)
        sStage[wid][r + 8 * hi][col] = (_Float16)silu_f(acc[j][r] + bb);
    }
    __builtin_amdgcn_wave_barrier();
    asm volatile("" ::: "memory");

    v8f acc2[4] = {};
#pragma unroll
    for (int kt = 0; kt < 4; ++kt) {
      v16h a;
#pragma unroll
      for (int v = 0; v < 8; ++v) {
        const int k = kt * 32 + kidx0(v, hi);
        a[2 * v]     = sStage[wid][m][k];
        a[2 * v + 1] = sStage[wid][m][k + 1];
      }
      mma_bcast<4, H_>(a, kt * 32, m, hi, &sW2[0][0], acc2);
    }

#pragma unroll
    for (int r = 0; r < 8; ++r) {
      const int nodeR = t * 16 + r + 8 * hi;
      if (nodeR < Nn) {
#pragma unroll
        for (int j = 0; j < 4; ++j) {
          const int col = j * 16 + m;
          out[(size_t)nodeR * DOUT_ + col] = acc2[j][r] + sB2[col];
        }
      }
    }
  }
}

// ---------------------------------------------------------------------------
extern "C" void kernel_launch(void* const* d_in, const int* in_sizes, int n_in,
                              void* d_out, int out_size, void* d_ws, size_t ws_size,
                              hipStream_t stream) {
  const float*     nf   = (const float*)d_in[0];
  const long long* eidx = (const long long*)d_in[1];
  const float*     ef   = (const float*)d_in[2];
  const float*     We1  = (const float*)d_in[3];
  const float*     be1  = (const float*)d_in[4];
  const float*     We2  = (const float*)d_in[5];
  const float*     be2  = (const float*)d_in[6];
  const float*     Wa   = (const float*)d_in[7];
  const float*     ba   = (const float*)d_in[8];
  const float*     Wn1  = (const float*)d_in[9];
  const float*     bn1  = (const float*)d_in[10];
  const float*     Wn2  = (const float*)d_in[11];
  const float*     bn2  = (const float*)d_in[12];

  const int Nn = in_sizes[0] / D_;
  const int E  = in_sizes[1] / 2;

  // Workspace layout: [agg f32 Nn*DOUT][nf16 f16 Nn*D]
  float*     agg  = (float*)d_ws;
  _Float16*  nf16 = (_Float16*)((char*)d_ws + (size_t)Nn * DOUT_ * sizeof(float));
  float*     out  = (float*)d_out;
  (void)out;

  // Tuple passthrough: (nodes, edge_index, edge_features) flattened.
  char* ob = (char*)d_out;
  const size_t nodesB = (size_t)Nn * DOUT_ * sizeof(float);
  const size_t eiB    = (size_t)2 * E * sizeof(long long);
  const size_t efB    = (size_t)E * DE_ * sizeof(float);
  hipMemcpyAsync(ob + nodesB, d_in[1], eiB, hipMemcpyDeviceToDevice, stream);
  hipMemcpyAsync(ob + nodesB + eiB, d_in[2], efB, hipMemcpyDeviceToDevice, stream);

  hipMemsetAsync(agg, 0, (size_t)Nn * DOUT_ * sizeof(float), stream);

  const int totalNF = Nn * D_;
  cvt_f16_kernel<<<(totalNF / 4 + 255) / 256, 256, 0, stream>>>(nf, nf16, totalNF);

  const int tilesE = (E + 15) / 16;
  int blocksE = (tilesE + WPB - 1) / WPB;
  if (blocksE > 16384) blocksE = 16384;
  edge_mlp_kernel<<<blocksE, 256, 0, stream>>>(nf16, eidx, ef, We1, be1, We2, be2,
                                               Wa, ba, agg, Nn, E);

  const int tilesN = (Nn + 15) / 16;
  int blocksN = (tilesN + WPB - 1) / WPB;
  if (blocksN > 16384) blocksN = 16384;
  node_mlp_kernel<<<blocksN, 256, 0, stream>>>(nf16, agg, Wn1, bn1, Wn2, bn2, out, Nn);
}
